// MultiHeadLatentAttention_16226386444620
// MI455X (gfx1250) — matrix operand order repaired and checked by reading, not compiled, not run on hardware
//
#include <hip/hip_runtime.h>
#include <hip/hip_bf16.h>
#include <stdint.h>

#define BB 2
#define TT 2048
#define DD 1024
#define HH 16
#define DHH 64
#define CC 128

typedef __bf16 bf16_t;
typedef __attribute__((ext_vector_type(16))) __bf16 v16bf;
typedef __attribute__((ext_vector_type(8)))  __bf16 v8bf;
typedef __attribute__((ext_vector_type(8)))  float  v8f;

// ---------------------------------------------------------------------------
// WMMA helpers (CDNA5 wave32, v_wmma_f32_16x16x32_bf16)
// ---------------------------------------------------------------------------
__device__ __forceinline__ v8f wmma_bf16(v16bf a, v16bf b, v8f c) {
  return __builtin_amdgcn_wmma_f32_16x16x32_bf16(
      /*neg_a=*/false, a, /*neg_b=*/false, b,
      /*c_mod=*/(short)0, c, /*reuse_a=*/false, /*reuse_b=*/false);
}

// A fragment (16x32, row-major A, lda in elements).
// ISA layout: lane 0-15 -> M=lane, K groups {0..7,16..23}; lanes 16-31 same M,
// K groups {8..15,24..31}.  Two 16B loads per lane.
__device__ __forceinline__ v16bf load_a_frag(const bf16_t* A, int lda,
                                             int m0, int k0, int lane) {
  int r = lane & 15, half = lane >> 4;
  const bf16_t* row = A + (size_t)(m0 + r) * lda + k0 + 8 * half;
  union { v16bf v; v8bf h[2]; } u;
  u.h[0] = *(const v8bf*)(row);
  u.h[1] = *(const v8bf*)(row + 16);
  return u.v;
}

// B fragment (32x16) read from BT = B^T stored [N][K] row-major.
// ISA layout: lanes 0-15 (N=lane) hold K {0..7,16..23}, lanes 16-31 hold K {8..15,24..31}.
// Two 16B reads per lane.
__device__ __forceinline__ v16bf load_b_frag(const bf16_t* BT, int ldb,
                                             int n0, int k0, int lane) {
  int c = lane & 15, half = lane >> 4;
  const bf16_t* row = BT + (size_t)(n0 + c) * ldb + k0 + 8 * half;
  union { v16bf v; v8bf h[2]; } u;
  u.h[0] = *(const v8bf*)(row);
  u.h[1] = *(const v8bf*)(row + 16);
  return u.v;
}

// A fragment from LDS (P matrix staging), row stride 72 elements (144B, 16B-aligned).
__device__ __forceinline__ v16bf load_a_frag_lds(const bf16_t* pl, int k0, int lane) {
  int r = lane & 15, half = lane >> 4;
  const bf16_t* row = pl + r * 72 + k0 + 8 * half;
  union { v16bf v; v8bf h[2]; } u;
  u.h[0] = *(const v8bf*)(row);
  u.h[1] = *(const v8bf*)(row + 16);
  return u.v;
}

// ---------------------------------------------------------------------------
// Elementwise conversion kernels
// ---------------------------------------------------------------------------
__global__ void cvt_bf16(const float* __restrict__ in, bf16_t* __restrict__ out, int n) {
  for (int i = blockIdx.x * blockDim.x + threadIdx.x; i < n; i += gridDim.x * blockDim.x)
    out[i] = (bf16_t)in[i];
}

// in: R x C row-major (f32)  ->  out: C x R row-major (bf16)   (i.e. B^T)
__global__ void cvt_transpose_bf16(const float* __restrict__ in, bf16_t* __restrict__ out,
                                   int R, int C) {
  int n = R * C;
  for (int i = blockIdx.x * blockDim.x + threadIdx.x; i < n; i += gridDim.x * blockDim.x) {
    int r = i / C, c = i - r * C;
    out[(size_t)c * R + r] = (bf16_t)in[i];
  }
}

// ---------------------------------------------------------------------------
// Generic WMMA GEMM: C[M][N] = A_bf[M][K] * BT_bf[N][K]^T + bias[N]
// One wave computes a 32x64 slab: 2 A-frags x 4 B-frags = 8 WMMAs / k-step.
// K loop is software-pipelined: next k-step's fragments are loaded before the
// current k-step's WMMAs issue, so loads overlap matrix math.
// Output: bf16 (Cb) or f32 (Cf).
// ---------------------------------------------------------------------------
__global__ __launch_bounds__(256)
void gemm_wmma_bf16(const bf16_t* __restrict__ A, const bf16_t* __restrict__ BT,
                    const float* __restrict__ bias,
                    bf16_t* __restrict__ Cb, float* __restrict__ Cf,
                    int M, int N, int K) {
  int lane = threadIdx.x & 31;
  int wave = blockIdx.x * (blockDim.x >> 5) + (threadIdx.x >> 5);
  int ntiles = N >> 6;                       // 64-wide slabs
  int total = (M >> 5) * ntiles;             // 32-row slabs
  if (wave >= total) return;
  int mt = wave / ntiles, nt = wave - mt * ntiles;
  int m0 = mt << 5, n0 = nt << 6;

  v8f acc[2][4] = {};
  v16bf a0, a1, b0, b1, b2, b3;

  // prologue: k-step 0
  a0 = load_a_frag(A, K, m0,      0, lane);
  a1 = load_a_frag(A, K, m0 + 16, 0, lane);
  b0 = load_b_frag(BT, K, n0,      0, lane);
  b1 = load_b_frag(BT, K, n0 + 16, 0, lane);
  b2 = load_b_frag(BT, K, n0 + 32, 0, lane);
  b3 = load_b_frag(BT, K, n0 + 48, 0, lane);

  for (int k0 = 32; k0 < K; k0 += 32) {
    // prefetch next k-step while current WMMAs execute
    v16bf na0 = load_a_frag(A, K, m0,      k0, lane);
    v16bf na1 = load_a_frag(A, K, m0 + 16, k0, lane);
    v16bf nb0 = load_b_frag(BT, K, n0,      k0, lane);
    v16bf nb1 = load_b_frag(BT, K, n0 + 16, k0, lane);
    v16bf nb2 = load_b_frag(BT, K, n0 + 32, k0, lane);
    v16bf nb3 = load_b_frag(BT, K, n0 + 48, k0, lane);

    acc[0][0] = wmma_bf16(a0, b0, acc[0][0]);
    acc[1][0] = wmma_bf16(a1, b0, acc[1][0]);
    acc[0][1] = wmma_bf16(a0, b1, acc[0][1]);
    acc[1][1] = wmma_bf16(a1, b1, acc[1][1]);
    acc[0][2] = wmma_bf16(a0, b2, acc[0][2]);
    acc[1][2] = wmma_bf16(a1, b2, acc[1][2]);
    acc[0][3] = wmma_bf16(a0, b3, acc[0][3]);
    acc[1][3] = wmma_bf16(a1, b3, acc[1][3]);

    a0 = na0; a1 = na1; b0 = nb0; b1 = nb1; b2 = nb2; b3 = nb3;
  }
  // tail k-step
  acc[0][0] = wmma_bf16(a0, b0, acc[0][0]);
  acc[1][0] = wmma_bf16(a1, b0, acc[1][0]);
  acc[0][1] = wmma_bf16(a0, b1, acc[0][1]);
  acc[1][1] = wmma_bf16(a1, b1, acc[1][1]);
  acc[0][2] = wmma_bf16(a0, b2, acc[0][2]);
  acc[1][2] = wmma_bf16(a1, b2, acc[1][2]);
  acc[0][3] = wmma_bf16(a0, b3, acc[0][3]);
  acc[1][3] = wmma_bf16(a1, b3, acc[1][3]);

  int half = lane >> 4, cl = lane & 15;
#pragma unroll
  for (int i = 0; i < 2; ++i) {
#pragma unroll
    for (int j = 0; j < 4; ++j) {
      int col = n0 + 16 * j + cl;
      float bv = bias ? bias[col] : 0.0f;
#pragma unroll
      for (int r = 0; r < 8; ++r) {
        int row = m0 + 16 * i + r + 8 * half;
        float v = acc[i][j][r] + bv;
        if (Cb) Cb[(size_t)row * N + col] = (bf16_t)v;
        else    Cf[(size_t)row * N + col] = v;
      }
    }
  }
}

// ---------------------------------------------------------------------------
// RoPE: q from x (f32), k from k_raw (bf16, [b*T+t][D]) -> [b][h][t][dh] bf16
// Thread handles pair (d, d+32) which share cos/sin.
// ---------------------------------------------------------------------------
__global__ void rope_qk(const float* __restrict__ x, const bf16_t* __restrict__ k_raw,
                        bf16_t* __restrict__ q, bf16_t* __restrict__ k) {
  int total = BB * HH * TT * 32;
  for (int idx = blockIdx.x * blockDim.x + threadIdx.x; idx < total;
       idx += gridDim.x * blockDim.x) {
    int j = idx & 31;
    int t = (idx >> 5) & (TT - 1);
    int h = (idx >> 16) & (HH - 1);
    int b = idx >> 20;
    float inv = __powf(10000.0f, -(float)j * (1.0f / 32.0f));
    float ang = (float)t * inv;
    float c = cosf(ang), s = sinf(ang);
    const float* xr = x + ((size_t)b * TT + t) * DD + h * DHH;
    float x0 = xr[j], x1 = xr[j + 32];
    size_t qo = (((size_t)b * HH + h) * TT + t) * DHH;
    q[qo + j]      = (bf16_t)(x0 * c - x1 * s);
    q[qo + j + 32] = (bf16_t)(x1 * c + x0 * s);
    const bf16_t* kr = k_raw + ((size_t)b * TT + t) * DD + h * DHH;
    float k0 = (float)kr[j], k1 = (float)kr[j + 32];
    k[qo + j]      = (bf16_t)(k0 * c - k1 * s);
    k[qo + j + 32] = (bf16_t)(k1 * c + k0 * s);
  }
}

// v_raw [b*T+t][D] (bf16) -> vT [b][h][dh][T] (bf16)
__global__ void transpose_v(const bf16_t* __restrict__ v_raw, bf16_t* __restrict__ vT) {
  int total = BB * TT * DD;
  for (int idx = blockIdx.x * blockDim.x + threadIdx.x; idx < total;
       idx += gridDim.x * blockDim.x) {
    int d = idx & (DD - 1);
    int t = (idx >> 10) & (TT - 1);
    int b = idx >> 21;
    int h = d >> 6, dh = d & 63;
    vT[(((size_t)b * HH + h) * DHH + dh) * TT + t] = v_raw[idx];
  }
}

// ---------------------------------------------------------------------------
// Flash-style attention. 4 waves / block, each wave owns 16 query rows of one
// (b,h); streams kv in 64-column chunks.  S and PV both via WMMA bf16.
// K-fragment loads are batched ahead of the S-WMMAs; V-fragment loads are
// hoisted before the softmax so their latency hides behind exp/shuffle work.
// q,k: [b][h][t][dh] bf16 ; vT: [b][h][dh][T] bf16
// out: [b][t][h*64+dh] bf16 (row-major (B*T) x D, feeds final GEMM directly)
// ---------------------------------------------------------------------------
__global__ __launch_bounds__(128)
void mla_attention(const bf16_t* __restrict__ q, const bf16_t* __restrict__ k,
                   const bf16_t* __restrict__ vT, bf16_t* __restrict__ attn_out) {
  const int wid = threadIdx.x >> 5, lane = threadIdx.x & 31;
  const int rblocks = TT / 64;
  const int bh = blockIdx.x / rblocks;
  const int rb = blockIdx.x - bh * rblocks;
  const int b = bh / HH, h = bh - b * HH;
  const int m0 = rb * 64 + wid * 16;

  const bf16_t* qh = q  + (size_t)bh * TT * DHH;
  const bf16_t* kh = k  + (size_t)bh * TT * DHH;
  const bf16_t* vh = vT + (size_t)bh * DHH * TT;   // [dh][T]

  // Q fragments for this wave's 16 rows (K = 64 -> two 16x32 frags), kept live.
  v16bf qa0 = load_a_frag(qh, DHH, m0, 0,  lane);
  v16bf qa1 = load_a_frag(qh, DHH, m0, 32, lane);

  const int half = lane >> 4, cl = lane & 15;
  const float scale = 0.125f;  // 1/sqrt(64)

  float m_r[8], l_r[8];
  v8f o[4] = {};
#pragma unroll
  for (int r = 0; r < 8; ++r) { m_r[r] = -1e30f; l_r[r] = 0.0f; }

  __shared__ __align__(32) bf16_t p_lds[4][16][72];
  bf16_t* pl = &p_lds[wid][0][0];

  for (int kc = 0; kc < TT; kc += 64) {
    // ---- batched K-fragment loads (one wait), then 8 S-WMMAs ----
    v16bf kb[8];
#pragma unroll
    for (int j = 0; j < 4; ++j) {
      kb[2 * j]     = load_b_frag(kh, DHH, kc + 16 * j, 0,  lane);
      kb[2 * j + 1] = load_b_frag(kh, DHH, kc + 16 * j, 32, lane);
    }
    v8f s[4] = {};
#pragma unroll
    for (int j = 0; j < 4; ++j) {
      s[j] = wmma_bf16(qa0, kb[2 * j],     s[j]);
      s[j] = wmma_bf16(qa1, kb[2 * j + 1], s[j]);
    }

    // ---- hoist V-fragment loads: latency hides behind softmax ----
    v16bf vb[8];
#pragma unroll
    for (int n = 0; n < 4; ++n) {
      vb[2 * n]     = load_b_frag(vh, TT, n * 16, kc,      lane);
      vb[2 * n + 1] = load_b_frag(vh, TT, n * 16, kc + 32, lane);
    }

    // prefetch next chunk of K rows into cache (global_prefetch_b8)
    if (kc + 64 < TT) {
      __builtin_prefetch(kh + (size_t)(kc + 64 + lane) * DHH, 0, 3);
      __builtin_prefetch(kh + (size_t)(kc + 96 + lane) * DHH, 0, 3);
    }

    // ---- online softmax over the 64-wide chunk ----
    float mc[8];
#pragma unroll
    for (int r = 0; r < 8; ++r) {
      float v = s[0][r];
      v = fmaxf(v, s[1][r]); v = fmaxf(v, s[2][r]); v = fmaxf(v, s[3][r]);
      mc[r] = v * scale;
    }
#pragma unroll
    for (int m = 1; m <= 8; m <<= 1)
#pragma unroll
      for (int r = 0; r < 8; ++r)
        mc[r] = fmaxf(mc[r], __shfl_xor(mc[r], m, 32));

    float alpha[8], m_new[8], lc[8];
#pragma unroll
    for (int r = 0; r < 8; ++r) {
      m_new[r] = fmaxf(m_r[r], mc[r]);
      alpha[r] = __expf(m_r[r] - m_new[r]);
      lc[r] = 0.0f;
    }

    // p = exp(s*scale - m_new), stage to LDS in (row, col) layout
#pragma unroll
    for (int j = 0; j < 4; ++j) {
#pragma unroll
      for (int r = 0; r < 8; ++r) {
        float p = __expf(s[j][r] * scale - m_new[r]);
        lc[r] += p;
        pl[(r + 8 * half) * 72 + 16 * j + cl] = (bf16_t)p;
      }
    }
#pragma unroll
    for (int m = 1; m <= 8; m <<= 1)
#pragma unroll
      for (int r = 0; r < 8; ++r)
        lc[r] += __shfl_xor(lc[r], m, 32);

#pragma unroll
    for (int r = 0; r < 8; ++r) {
      l_r[r] = l_r[r] * alpha[r] + lc[r];
      m_r[r] = m_new[r];
    }
#pragma unroll
    for (int n = 0; n < 4; ++n)
#pragma unroll
      for (int r = 0; r < 8; ++r)
        o[n][r] *= alpha[r];

    __syncthreads();   // make wave's P visible (and order LDS ops)

    // ---- O += P . V for this chunk (8 WMMAs, V already in registers) ----
    v16bf pa0 = load_a_frag_lds(pl, 0,  lane);
    v16bf pa1 = load_a_frag_lds(pl, 32, lane);
#pragma unroll
    for (int n = 0; n < 4; ++n) {
      o[n] = wmma_bf16(pa0, vb[2 * n],     o[n]);
      o[n] = wmma_bf16(pa1, vb[2 * n + 1], o[n]);
    }
    __syncthreads();   // WAR protection before next chunk's P stores
  }

  // ---- epilogue: normalize, write [b][t][h*64+dh] bf16 ----
#pragma unroll
  for (int n = 0; n < 4; ++n) {
#pragma unroll
    for (int r = 0; r < 8; ++r) {
      int t = m0 + r + 8 * half;
      float val = o[n][r] / l_r[r];
      attn_out[((size_t)b * TT + t) * DD + h * DHH + n * 16 + cl] = (bf16_t)val;
    }
  }
}

// ---------------------------------------------------------------------------
// Host-side launch
// ---------------------------------------------------------------------------
extern "C" void kernel_launch(void* const* d_in, const int* in_sizes, int n_in,
                              void* d_out, int out_size, void* d_ws, size_t ws_size,
                              hipStream_t stream) {
  const float* x    = (const float*)d_in[0];
  const float* w_dk = (const float*)d_in[1];
  const float* b_dk = (const float*)d_in[2];
  const float* w_dv = (const float*)d_in[3];
  const float* b_dv = (const float*)d_in[4];
  const float* w_uk = (const float*)d_in[5];
  const float* b_uk = (const float*)d_in[6];
  const float* w_uv = (const float*)d_in[7];
  const float* b_uv = (const float*)d_in[8];
  const float* w_o  = (const float*)d_in[9];
  const float* b_o  = (const float*)d_in[10];
  float* out = (float*)d_out;

  const int M = BB * TT;           // 4096

  char* w = (char*)d_ws;
  auto alloc = [&](size_t bytes) -> void* {
    void* p = (void*)w;
    w += (bytes + 255) & ~(size_t)255;
    return p;
  };
  bf16_t* x_bf   = (bf16_t*)alloc((size_t)M * DD * 2);
  bf16_t* wdkT   = (bf16_t*)alloc((size_t)CC * DD * 2);
  bf16_t* wdvT   = (bf16_t*)alloc((size_t)CC * DD * 2);
  bf16_t* wukT   = (bf16_t*)alloc((size_t)DD * CC * 2);
  bf16_t* wuvT   = (bf16_t*)alloc((size_t)DD * CC * 2);
  bf16_t* woT    = (bf16_t*)alloc((size_t)DD * DD * 2);
  bf16_t* c_k    = (bf16_t*)alloc((size_t)M * CC * 2);
  bf16_t* c_v    = (bf16_t*)alloc((size_t)M * CC * 2);
  bf16_t* k_raw  = (bf16_t*)alloc((size_t)M * DD * 2);
  bf16_t* v_raw  = (bf16_t*)alloc((size_t)M * DD * 2);
  bf16_t* q_bf   = (bf16_t*)alloc((size_t)M * DD * 2);
  bf16_t* k_bf   = (bf16_t*)alloc((size_t)M * DD * 2);
  bf16_t* vT     = (bf16_t*)alloc((size_t)M * DD * 2);
  bf16_t* attn   = (bf16_t*)alloc((size_t)M * DD * 2);

  // 1) conversions
  cvt_bf16<<<2048, 256, 0, stream>>>(x, x_bf, M * DD);
  cvt_transpose_bf16<<<512, 256, 0, stream>>>(w_dk, wdkT, DD, CC);
  cvt_transpose_bf16<<<512, 256, 0, stream>>>(w_dv, wdvT, DD, CC);
  cvt_transpose_bf16<<<512, 256, 0, stream>>>(w_uk, wukT, CC, DD);
  cvt_transpose_bf16<<<512, 256, 0, stream>>>(w_uv, wuvT, CC, DD);
  cvt_transpose_bf16<<<2048, 256, 0, stream>>>(w_o, woT, DD, DD);

  auto gemm_blocks = [](int Mm, int Nn) { return ((Mm >> 5) * (Nn >> 6) + 7) / 8; };

  // 2) down-projections: (4096x1024) x (1024x128)
  gemm_wmma_bf16<<<gemm_blocks(M, CC), 256, 0, stream>>>(x_bf, wdkT, b_dk, c_k, nullptr, M, CC, DD);
  gemm_wmma_bf16<<<gemm_blocks(M, CC), 256, 0, stream>>>(x_bf, wdvT, b_dv, c_v, nullptr, M, CC, DD);

  // 3) up-projections: (4096x128) x (128x1024)
  gemm_wmma_bf16<<<gemm_blocks(M, DD), 256, 0, stream>>>(c_k, wukT, b_uk, k_raw, nullptr, M, DD, CC);
  gemm_wmma_bf16<<<gemm_blocks(M, DD), 256, 0, stream>>>(c_v, wuvT, b_uv, v_raw, nullptr, M, DD, CC);

  // 4) RoPE on q (from x) and k; per-head layouts
  rope_qk<<<4096, 256, 0, stream>>>(x, k_raw, q_bf, k_bf);

  // 5) v -> [b][h][dh][T]
  transpose_v<<<8192, 256, 0, stream>>>(v_raw, vT);

  // 6) attention: B*H * (T/64) blocks, 128 threads
  mla_attention<<<BB * HH * (TT / 64), 128, 0, stream>>>(q_bf, k_bf, vT, attn);

  // 7) output projection: (4096x1024) x (1024x1024) + b_o -> f32 d_out
  gemm_wmma_bf16<<<gemm_blocks(M, DD), 256, 0, stream>>>(attn, woT, b_o, nullptr, out, M, DD, DD);

  (void)in_sizes; (void)n_in; (void)out_size; (void)ws_size;
}
